// Model_26070451487349
// MI455X (gfx1250) — matrix operand order repaired and checked by reading, not compiled, not run on hardware
//
#include <hip/hip_runtime.h>

#define BB 8
#define L_ENC 720
#define L_ENC2 360
#define L_DEC 672
#define C_IN 321
#define DM 512
#define NH 8
#define DH 64
#define DFF 2048
#define PRED 336

typedef _Float16 half_t;
typedef __attribute__((ext_vector_type(16))) _Float16 v16h;
typedef __attribute__((ext_vector_type(8))) _Float16 v8h;
typedef __attribute__((ext_vector_type(8))) float v8f;

static inline int ceil_div(int a, int b) { return (a + b - 1) / b; }

__device__ __forceinline__ unsigned hash_u32(unsigned x) {
  x ^= x >> 16; x *= 0x7feb352dU; x ^= x >> 15; x *= 0x846ca68bU; x ^= x >> 16;
  return x;
}

// ---------------------------------------------------------------------------
// Instance-norm stats over time axis for x_enc; writes normalized xe.
// ---------------------------------------------------------------------------
__global__ void norm_stats_kernel(const float* __restrict__ x, float* __restrict__ means,
                                  float* __restrict__ stdev, float* __restrict__ xe) {
  int t = blockIdx.x * blockDim.x + threadIdx.x;
  if (t >= BB * C_IN) return;
  int b = t / C_IN, c = t % C_IN;
  const float* xb = x + (size_t)b * L_ENC * C_IN + c;
  float s = 0.f, s2 = 0.f;
  for (int l = 0; l < L_ENC; ++l) { float v = xb[(size_t)l * C_IN]; s += v; s2 += v * v; }
  float mu = s / (float)L_ENC;
  float var = s2 / (float)L_ENC - mu * mu;
  float sd = sqrtf(var + 1e-5f);
  means[t] = mu; stdev[t] = sd;
  float inv = 1.f / sd;
  float* xo = xe + (size_t)b * L_ENC * C_IN + c;
  for (int l = 0; l < L_ENC; ++l) xo[(size_t)l * C_IN] = (xb[(size_t)l * C_IN] - mu) * inv;
}

// ---------------------------------------------------------------------------
// Data embedding: circular conv3 over input channels + positional + mark proj
// ---------------------------------------------------------------------------
__global__ void embed_kernel(const float* __restrict__ x, const float* __restrict__ xmark,
                             const float* __restrict__ convw, const float* __restrict__ markw,
                             float* __restrict__ out, int L) {
  int t = blockIdx.x * blockDim.x + threadIdx.x;
  if (t >= BB * L * DM) return;
  int d = t % DM; int bl = t / DM; int l = bl % L; int b = bl / L;
  int lm1 = (l == 0) ? L - 1 : l - 1;
  int lp1 = (l == L - 1) ? 0 : l + 1;
  const float* x0 = x + ((size_t)b * L + lm1) * C_IN;
  const float* x1 = x + ((size_t)b * L + l) * C_IN;
  const float* x2 = x + ((size_t)b * L + lp1) * C_IN;
  const float* w = convw + (size_t)d * C_IN * 3;
  float acc = 0.f;
  for (int c = 0; c < C_IN; ++c)
    acc += x0[c] * w[c * 3 + 0] + x1[c] * w[c * 3 + 1] + x2[c] * w[c * 3 + 2];
  int i2 = d >> 1;
  float div = expf(-(float)(2 * i2) * (9.210340371976184f / (float)DM));
  float ang = (float)l * div;
  acc += (d & 1) ? cosf(ang) : sinf(ang);
  const float* mr = xmark + ((size_t)b * L + l) * 4;
  acc += mr[0] * markw[0 * DM + d] + mr[1] * markw[1 * DM + d] +
         mr[2] * markw[2 * DM + d] + mr[3] * markw[3 * DM + d];
  out[t] = acc;
}

// ---------------------------------------------------------------------------
// Weight repack: W[K,N] fp32 -> Wt[Npad,K] f16 (transposed, zero-padded cols).
// Amortized over M/16 tile reuses; makes B-fragment loads K-contiguous b128s.
// ---------------------------------------------------------------------------
__global__ void convert_wT_kernel(const float* __restrict__ W, half_t* __restrict__ Wt,
                                  int N, int Npad, int K) {
  int t = blockIdx.x * blockDim.x + threadIdx.x;
  if (t >= Npad * K) return;
  int k = t % K; int n = t / K;
  Wt[t] = (half_t)((n < N) ? W[(size_t)k * N + n] : 0.f);
}

// Distill-conv weight w[d,c,k] -> Wt[n=d][kk=c*3+k] is the identity linear
// index, so the repack is a pure fp32->f16 cast.
__global__ void castW_kernel(const float* __restrict__ w, half_t* __restrict__ wt, int n) {
  int t = blockIdx.x * blockDim.x + threadIdx.x;
  if (t < n) wt[t] = (half_t)w[t];
}

// ---------------------------------------------------------------------------
// Fragment loaders (ISA 7.12.2 f16 layouts)
// ---------------------------------------------------------------------------
__device__ __forceinline__ v16h loadA_frag(const float* __restrict__ Ar) {
  float4 a0 = *(const float4*)(Ar);
  float4 a1 = *(const float4*)(Ar + 4);
  float4 a2 = *(const float4*)(Ar + 16);
  float4 a3 = *(const float4*)(Ar + 20);
  v16h af;
  af[0] = (half_t)a0.x;  af[1] = (half_t)a0.y;  af[2] = (half_t)a0.z;  af[3] = (half_t)a0.w;
  af[4] = (half_t)a1.x;  af[5] = (half_t)a1.y;  af[6] = (half_t)a1.z;  af[7] = (half_t)a1.w;
  af[8] = (half_t)a2.x;  af[9] = (half_t)a2.y;  af[10] = (half_t)a2.z; af[11] = (half_t)a2.w;
  af[12] = (half_t)a3.x; af[13] = (half_t)a3.y; af[14] = (half_t)a3.z; af[15] = (half_t)a3.w;
  return af;
}

__device__ __forceinline__ v16h loadB_frag(const half_t* __restrict__ Wp) {
  v8h lo = *(const v8h*)(Wp);
  v8h hi = *(const v8h*)(Wp + 16);
  return __builtin_shufflevector(lo, hi, 0, 1, 2, 3, 4, 5, 6, 7,
                                 8, 9, 10, 11, 12, 13, 14, 15);
}

// ---------------------------------------------------------------------------
// WMMA GEMM: C[M,N] = act(A[M,K] @ W + bias) + res, W pre-packed as Wt[Npad,K].
// One 16x32 output tile per wave32: 1 A-fragment, 2 B-fragments, 2 WMMAs per
// K-step of 32. K % 32 == 0 and K >= 64 at every call site.
// Software-pipelined: next K-step's loads issue before the current WMMAs so
// the s_wait lands after the matrix ops instead of stalling them.
// ---------------------------------------------------------------------------
__global__ __launch_bounds__(256) void gemm_wmma_kernel(
    const float* __restrict__ A, const half_t* __restrict__ Wt,
    const float* __restrict__ bias, const float* __restrict__ res,
    float* __restrict__ C, int M, int N, int Npad, int K, int act) {
  const int lane = threadIdx.x & 31;
  const int wave = threadIdx.x >> 5;
  const int tilesN = Npad >> 5;
  int tile = blockIdx.x * 8 + wave;
  if (tile >= (M >> 4) * tilesN) return;
  const int tm = tile / tilesN;
  const int tn = tile - tm * tilesN;
  const int row0 = tm << 4, col0 = tn << 5;

  const int mrow = lane & 15;          // A row / B col within tile
  const int ka = (lane >> 4) << 3;     // A K-phase: 0 or 8
  const int kb = (lane >> 4) << 3;     // B K-phase: 0 or 8
  const float* Ar = A + (size_t)(row0 + mrow) * K + ka;
  const half_t* W0 = Wt + (size_t)(col0 + mrow) * K + kb;
  const half_t* W1 = W0 + (size_t)16 * K;

  v8f acc0 = {}, acc1 = {};
  v16h af = loadA_frag(Ar);
  v16h bf0 = loadB_frag(W0);
  v16h bf1 = loadB_frag(W1);
  for (int k0 = 32; k0 < K; k0 += 32) {
    v16h naf = loadA_frag(Ar + k0);
    v16h nbf0 = loadB_frag(W0 + k0);
    v16h nbf1 = loadB_frag(W1 + k0);
    acc0 = __builtin_amdgcn_wmma_f32_16x16x32_f16(false, af, false, bf0,
                                                  (short)0, acc0, false, false);
    acc1 = __builtin_amdgcn_wmma_f32_16x16x32_f16(false, af, false, bf1,
                                                  (short)0, acc1, false, false);
    af = naf; bf0 = nbf0; bf1 = nbf1;
  }
  acc0 = __builtin_amdgcn_wmma_f32_16x16x32_f16(false, af, false, bf0,
                                                (short)0, acc0, false, false);
  acc1 = __builtin_amdgcn_wmma_f32_16x16x32_f16(false, af, false, bf1,
                                                (short)0, acc1, false, false);

  const int mbase = (lane >> 4) << 3;  // C: vgpr i -> row mbase+i
  int nc0 = col0 + (lane & 15);
  auto emit = [&](v8f& acc, int nc) {
    if (nc >= N) return;
    float bv = bias ? bias[nc] : 0.f;
#pragma unroll
    for (int i = 0; i < 8; ++i) {
      int r = row0 + mbase + i;
      float v = acc[i] + bv;
      if (act == 1) v = 0.5f * v * (1.f + erff(v * 0.7071067811865476f));
      if (res) v += res[(size_t)r * N + nc];
      C[(size_t)r * N + nc] = v;
    }
  };
  emit(acc0, nc0);
  emit(acc1, nc0 + 16);
}

// ---------------------------------------------------------------------------
// LayerNorm over last dim (DM), one wave per row.
// ---------------------------------------------------------------------------
__global__ void ln_kernel(const float* __restrict__ x, const float* __restrict__ g,
                          const float* __restrict__ b, float* __restrict__ out, int rows) {
  int row = blockIdx.x * (blockDim.x >> 5) + (threadIdx.x >> 5);
  if (row >= rows) return;
  int lane = threadIdx.x & 31;
  const float* xr = x + (size_t)row * DM;
  float s = 0.f, s2 = 0.f;
  for (int i = lane; i < DM; i += 32) { float v = xr[i]; s += v; s2 += v * v; }
  for (int m = 16; m >= 1; m >>= 1) { s += __shfl_xor(s, m, 32); s2 += __shfl_xor(s2, m, 32); }
  float mu = s / (float)DM;
  float var = s2 / (float)DM - mu * mu;
  float inv = rsqrtf(var + 1e-5f);
  float* orow = out + (size_t)row * DM;
  for (int i = lane; i < DM; i += 32) orow[i] = (xr[i] - mu) * inv * g[i] + b[i];
}

// ---------------------------------------------------------------------------
// ProbSparse attention pieces. q/k/v/ctx stored [B, L, H*DH] row-major.
// ---------------------------------------------------------------------------
__global__ void ctx_mean_kernel(const float* __restrict__ v, float* __restrict__ ctx,
                                int Lq, int Lk) {
  int t = blockIdx.x * blockDim.x + threadIdx.x;
  if (t >= BB * NH * DH) return;
  int d = t % DH; int bh = t / DH; int h = bh % NH; int b = bh / NH;
  size_t vb = (size_t)b * Lk * DM + h * DH + d;
  float s = 0.f;
  for (int j = 0; j < Lk; ++j) s += v[vb + (size_t)j * DM];
  s /= (float)Lk;
  size_t cb = (size_t)b * Lq * DM + h * DH + d;
  for (int l = 0; l < Lq; ++l) ctx[cb + (size_t)l * DM] = s;
}

__global__ void ctx_cumsum_kernel(const float* __restrict__ v, float* __restrict__ ctx, int L) {
  int t = blockIdx.x * blockDim.x + threadIdx.x;
  if (t >= BB * NH * DH) return;
  int d = t % DH; int bh = t / DH; int h = bh % NH; int b = bh / NH;
  size_t base = (size_t)b * L * DM + h * DH + d;
  float s = 0.f;
  for (int j = 0; j < L; ++j) { s += v[base + (size_t)j * DM]; ctx[base + (size_t)j * DM] = s; }
}

__global__ void sampleM_kernel(const float* __restrict__ q, const float* __restrict__ k,
                               float* __restrict__ Mout, int Lq, int Lk, int U, unsigned seed) {
  int t = blockIdx.x * blockDim.x + threadIdx.x;
  if (t >= BB * NH * Lq) return;
  int lq = t % Lq; int bh = t / Lq; int h = bh % NH; int b = bh / NH;
  const float* qr = q + ((size_t)b * Lq + lq) * DM + h * DH;
  float mx = -3.4e38f, sm = 0.f;
  for (int j = 0; j < U; ++j) {
    unsigned r = hash_u32(seed * 0x9E3779B9u + (unsigned)(lq * U + j) * 2654435761u);
    int idx = (int)(r % (unsigned)Lk);
    const float* kr = k + ((size_t)b * Lk + idx) * DM + h * DH;
    float s = 0.f;
    for (int d = 0; d < DH; ++d) s += qr[d] * kr[d];
    mx = fmaxf(mx, s); sm += s;
  }
  Mout[(size_t)bh * Lq + lq] = mx - sm / (float)Lk;
}

__global__ void topk_kernel(float* __restrict__ M, int* __restrict__ top, int Lq, int u) {
  int t = threadIdx.x;
  if (t >= BB * NH) return;
  float* Mr = M + (size_t)t * Lq;
  int* tp = top + (size_t)t * 64;
  for (int s = 0; s < u; ++s) {
    float best = -3.4e38f; int bi = 0;
    for (int l = 0; l < Lq; ++l)
      if (Mr[l] > best) { best = Mr[l]; bi = l; }
    tp[s] = bi;
    Mr[bi] = -3.4e38f;
  }
}

__global__ void sparse_attn_kernel(const float* __restrict__ q, const float* __restrict__ k,
                                   const float* __restrict__ v, const int* __restrict__ top,
                                   float* __restrict__ ctx, int Lq, int Lk, int u, int causal) {
  __shared__ float sc[L_ENC];  // >= max Lk (720)
  int ui = blockIdx.x % u; int bh = blockIdx.x / u; int h = bh % NH; int b = bh / NH;
  int lq = top[(size_t)bh * 64 + ui];
  int lane = threadIdx.x;
  const float* qr = q + ((size_t)b * Lq + lq) * DM + h * DH;
  for (int j = lane; j < Lk; j += 32) {
    const float* kr = k + ((size_t)b * Lk + j) * DM + h * DH;
    float s = 0.f;
    for (int d = 0; d < DH; ++d) s += qr[d] * kr[d];
    s *= 0.125f;  // 1/sqrt(64)
    if (causal && j > lq) s = -3.4e38f;
    sc[j] = s;
  }
  __syncthreads();
  float mx = -3.4e38f;
  for (int j = lane; j < Lk; j += 32) mx = fmaxf(mx, sc[j]);
  for (int m = 16; m >= 1; m >>= 1) mx = fmaxf(mx, __shfl_xor(mx, m, 32));
  float ss = 0.f;
  for (int j = lane; j < Lk; j += 32) { float e = expf(sc[j] - mx); sc[j] = e; ss += e; }
  for (int m = 16; m >= 1; m >>= 1) ss += __shfl_xor(ss, m, 32);
  __syncthreads();
  float inv = 1.f / ss;
  for (int d = lane; d < DH; d += 32) {
    float a = 0.f;
    for (int j = 0; j < Lk; ++j) a += sc[j] * v[((size_t)b * Lk + j) * DM + h * DH + d];
    ctx[((size_t)b * Lq + lq) * DM + h * DH + d] = a * inv;
  }
}

// ---------------------------------------------------------------------------
// Distilling conv helpers: im2col (circular, tap 3) + BN/ELU/maxpool
// ---------------------------------------------------------------------------
__global__ void im2col3_kernel(const float* __restrict__ x, float* __restrict__ out, int L) {
  int t = blockIdx.x * blockDim.x + threadIdx.x;
  if (t >= BB * L * DM * 3) return;
  int k = t % 3; int c = (t / 3) % DM; int bl = t / (3 * DM); int l = bl % L; int b = bl / L;
  int ls = l + k - 1;
  ls = (ls < 0) ? (L - 1) : ((ls >= L) ? 0 : ls);
  out[t] = x[((size_t)b * L + ls) * DM + c];
}

__global__ void bn_elu_pool_kernel(const float* __restrict__ x, const float* __restrict__ g,
                                   const float* __restrict__ bb, float* __restrict__ out,
                                   int L, int Lo) {
  int t = blockIdx.x * blockDim.x + threadIdx.x;
  if (t >= BB * Lo * DM) return;
  int d = t % DM; int bl = t / DM; int lo = bl % Lo; int b = bl / Lo;
  float scale = g[d] * rsqrtf(1.f + 1e-5f);
  float beta = bb[d];
  float best = -3.4e38f;
  for (int tk = 0; tk < 3; ++tk) {
    int li = 2 * lo - 1 + tk;
    if (li < 0 || li >= L) continue;
    float y = x[((size_t)b * L + li) * DM + d] * scale + beta;
    y = (y > 0.f) ? y : expm1f(y);
    best = fmaxf(best, y);
  }
  out[t] = best;
}

// ---------------------------------------------------------------------------
// De-normalize first PRED rows of decoder projection into d_out.
// ---------------------------------------------------------------------------
__global__ void final_kernel(const float* __restrict__ p, const float* __restrict__ stdev,
                             const float* __restrict__ means, float* __restrict__ out) {
  int t = blockIdx.x * blockDim.x + threadIdx.x;
  if (t >= BB * PRED * C_IN) return;
  int c = t % C_IN; int bt = t / C_IN; int tt = bt % PRED; int b = bt / PRED;
  float y = p[((size_t)b * L_DEC + tt) * C_IN + c];
  out[t] = y * stdev[b * C_IN + c] + means[b * C_IN + c];
}

// ---------------------------------------------------------------------------
// Host-side launchers
// ---------------------------------------------------------------------------
static inline void gemm_core(hipStream_t s, const float* A, const half_t* Wt,
                             const float* bias, const float* res, float* C,
                             int M, int N, int Npad, int K, int act) {
  int tiles = (M / 16) * (Npad / 32);
  gemm_wmma_kernel<<<ceil_div(tiles, 8), 256, 0, s>>>(A, Wt, bias, res, C, M, N, Npad, K, act);
}

static inline void gemm(hipStream_t s, const float* A, const float* W, const float* bias,
                        const float* res, float* C, int M, int N, int K, int act, half_t* wt) {
  int Npad = ceil_div(N, 32) * 32;
  convert_wT_kernel<<<ceil_div(Npad * K, 256), 256, 0, s>>>(W, wt, N, Npad, K);
  gemm_core(s, A, wt, bias, res, C, M, N, Npad, K, act);
}

static inline void layernorm(hipStream_t s, const float* x, const float* g, const float* b,
                             float* out, int rows) {
  ln_kernel<<<ceil_div(rows, 8), 256, 0, s>>>(x, g, b, out, rows);
}

static inline void prob_attn(hipStream_t s, const float* q, const float* k, const float* v,
                             float* ctx, int Lq, int Lk, int U, int u, int causal,
                             unsigned seed, float* Mbuf, int* topbuf) {
  if (causal)
    ctx_cumsum_kernel<<<ceil_div(BB * NH * DH, 256), 256, 0, s>>>(v, ctx, Lq);
  else
    ctx_mean_kernel<<<ceil_div(BB * NH * DH, 256), 256, 0, s>>>(v, ctx, Lq, Lk);
  sampleM_kernel<<<ceil_div(BB * NH * Lq, 256), 256, 0, s>>>(q, k, Mbuf, Lq, Lk, U, seed);
  topk_kernel<<<1, BB * NH, 0, s>>>(Mbuf, topbuf, Lq, u);
  sparse_attn_kernel<<<BB * NH * u, 32, 0, s>>>(q, k, v, topbuf, ctx, Lq, Lk, u, causal);
}

extern "C" void kernel_launch(void* const* d_in, const int* in_sizes, int n_in,
                              void* d_out, int out_size, void* d_ws, size_t ws_size,
                              hipStream_t stream) {
  auto in = [&](int i) { return (const float*)d_in[i]; };

  // Workspace bump allocator (floats)
  float* wsf = (float*)d_ws;
  size_t off = 0;
  auto alloc = [&](size_t n) { float* p = wsf + off; off += n; return p; };
  float* means = alloc((size_t)BB * C_IN);
  float* stdev = alloc((size_t)BB * C_IN);
  float* xe = alloc((size_t)BB * L_ENC * C_IN);
  float* X = alloc((size_t)BB * L_ENC * DM);   // encoder activation
  float* Y = alloc((size_t)BB * L_DEC * DM);   // decoder activation
  float* Q = alloc((size_t)BB * L_ENC * DM);
  float* Kb = alloc((size_t)BB * L_ENC * DM);
  float* Vb = alloc((size_t)BB * L_ENC * DM);
  float* CT = alloc((size_t)BB * L_ENC * DM);
  float* T0 = alloc((size_t)BB * L_ENC * DM);
  float* Hb = alloc((size_t)BB * L_ENC * DFF); // FFN hidden, reused as im2col
  float* CR = alloc((size_t)BB * L_ENC2 * DM); // encoder output (cross)
  half_t* Wt = (half_t*)alloc(600000);         // packed-weight arena (>= 1M halfs)
  float* Pb = alloc((size_t)BB * L_DEC * C_IN);
  float* Mb = alloc((size_t)BB * NH * L_ENC);
  int* top = (int*)(wsf + off); off += (size_t)BB * NH * 64;
  (void)ws_size; (void)in_sizes; (void)n_in; (void)out_size;

  // ---- normalize + encoder embedding ----
  norm_stats_kernel<<<ceil_div(BB * C_IN, 256), 256, 0, stream>>>(in(0), means, stdev, xe);
  embed_kernel<<<ceil_div(BB * L_ENC * DM, 256), 256, 0, stream>>>(
      xe, in(1), in(4), in(5), X, L_ENC);

  // ---- encoder: 2 layers, distill conv between them ----
  const int encL[2] = {L_ENC, L_ENC2};
  const int encBase[2] = {8, 24};
  const int encU[2] = {35, 30};  // 5*ceil(ln(720))=35, 5*ceil(ln(360))=30
  for (int li = 0; li < 2; ++li) {
    int L = encL[li]; int M = BB * L; int pb = encBase[li];
    gemm(stream, X, in(pb + 0), in(pb + 1), nullptr, Q, M, DM, DM, 0, Wt);
    gemm(stream, X, in(pb + 2), in(pb + 3), nullptr, Kb, M, DM, DM, 0, Wt);
    gemm(stream, X, in(pb + 4), in(pb + 5), nullptr, Vb, M, DM, DM, 0, Wt);
    prob_attn(stream, Q, Kb, Vb, CT, L, L, encU[li], encU[li], 0, 1u + li, Mb, top);
    gemm(stream, CT, in(pb + 6), in(pb + 7), X, T0, M, DM, DM, 0, Wt);  // + residual
    layernorm(stream, T0, in(pb + 8), in(pb + 9), X, M);
    gemm(stream, X, in(pb + 12), in(pb + 13), nullptr, Hb, M, DFF, DM, 1, Wt);  // GELU
    gemm(stream, Hb, in(pb + 14), in(pb + 15), X, T0, M, DM, DFF, 0, Wt);
    layernorm(stream, T0, in(pb + 10), in(pb + 11), X, M);
    if (li == 0) {
      // conv weight w[d,c,k] -> Wt[d][c*3+k] is identity indexing: pure cast
      castW_kernel<<<ceil_div(DM * DM * 3, 256), 256, 0, stream>>>(in(40), Wt, DM * DM * 3);
      im2col3_kernel<<<ceil_div(BB * L * DM * 3, 256), 256, 0, stream>>>(X, Hb, L);
      gemm_core(stream, Hb, Wt, in(41), nullptr, T0, M, DM, DM, DM * 3, 0);
      bn_elu_pool_kernel<<<ceil_div(BB * L_ENC2 * DM, 256), 256, 0, stream>>>(
          T0, in(42), in(43), X, L, L_ENC2);
    }
  }
  layernorm(stream, X, in(44), in(45), CR, BB * L_ENC2);

  // ---- decoder embedding ----
  embed_kernel<<<ceil_div(BB * L_DEC * DM, 256), 256, 0, stream>>>(
      in(2), in(3), in(6), in(7), Y, L_DEC);

  int Md = BB * L_DEC;
  // self-attention (causal, cumsum context); 5*ceil(ln(672))=35
  gemm(stream, Y, in(46), in(47), nullptr, Q, Md, DM, DM, 0, Wt);
  gemm(stream, Y, in(48), in(49), nullptr, Kb, Md, DM, DM, 0, Wt);
  gemm(stream, Y, in(50), in(51), nullptr, Vb, Md, DM, DM, 0, Wt);
  prob_attn(stream, Q, Kb, Vb, CT, L_DEC, L_DEC, 35, 35, 1, 100u, Mb, top);
  gemm(stream, CT, in(52), in(53), Y, T0, Md, DM, DM, 0, Wt);
  layernorm(stream, T0, in(62), in(63), Y, Md);

  // cross-attention: U from Lk=360 -> 30, u from Lq=672 -> 35
  gemm(stream, Y, in(54), in(55), nullptr, Q, Md, DM, DM, 0, Wt);
  gemm(stream, CR, in(56), in(57), nullptr, Kb, BB * L_ENC2, DM, DM, 0, Wt);
  gemm(stream, CR, in(58), in(59), nullptr, Vb, BB * L_ENC2, DM, DM, 0, Wt);
  prob_attn(stream, Q, Kb, Vb, CT, L_DEC, L_ENC2, 30, 35, 0, 200u, Mb, top);
  gemm(stream, CT, in(60), in(61), Y, T0, Md, DM, DM, 0, Wt);
  layernorm(stream, T0, in(64), in(65), Y, Md);

  // decoder FFN
  gemm(stream, Y, in(68), in(69), nullptr, Hb, Md, DFF, DM, 1, Wt);
  gemm(stream, Hb, in(70), in(71), Y, T0, Md, DM, DFF, 0, Wt);
  layernorm(stream, T0, in(66), in(67), Y, Md);

  // final norm + projection + de-normalize
  layernorm(stream, Y, in(72), in(73), T0, Md);
  gemm(stream, T0, in(74), in(75), nullptr, Pb, Md, C_IN, DM, 0, Wt);
  final_kernel<<<ceil_div(BB * PRED * C_IN, 256), 256, 0, stream>>>(
      Pb, stdev, means, (float*)d_out);
}
